// Transformer_67877663146116
// MI455X (gfx1250) — matrix operand order repaired and checked by reading, not compiled, not run on hardware
//
#include <hip/hip_runtime.h>
#include <math.h>

// ---------------- CDNA5 WMMA types ----------------
typedef __attribute__((ext_vector_type(16))) __bf16 v16bf;
typedef __attribute__((ext_vector_type(4)))  __bf16 v4bf;
typedef __attribute__((ext_vector_type(8)))  float  v8f;
typedef __attribute__((ext_vector_type(4)))  int    i32x4;

#if defined(__AMDGCN__) && __has_builtin(__builtin_amdgcn_global_load_async_to_lds_b128)
#define USE_ASYNC_LDS 1
#else
#define USE_ASYNC_LDS 0
#endif

__device__ __forceinline__ void wait_async_lds() {
#if defined(__AMDGCN__)
#if __has_builtin(__builtin_amdgcn_s_wait_asynccnt)
  __builtin_amdgcn_s_wait_asynccnt(0);
#else
  asm volatile("s_wait_asynccnt 0x0" ::: "memory");
#endif
#endif
}

__device__ __forceinline__ v8f wmma_bf16(v16bf a, v16bf b, v8f c) {
  // D = A(16x32 bf16) * B(32x16 bf16) + C(16x16 f32)
  return __builtin_amdgcn_wmma_f32_16x16x32_bf16(
      /*neg_a=*/false, a, /*neg_b=*/false, b,
      /*c_mod=*/(short)0, c, /*reuse_a=*/false, /*reuse_b=*/false);
}

// A-fragment element->K mapping (16-bit A, 16x32, wave32)
__device__ __forceinline__ int a_kidx(int e, int half) {
  return (e < 8) ? (half * 8 + e) : (16 + half * 8 + (e - 8));
}
// B-fragment element->K mapping (16-bit B, 32x16): same as A
__device__ __forceinline__ int b_kidx(int e, int half) { return a_kidx(e, half); }

__device__ __forceinline__ float sigmoidf_(float x) { return 1.0f / (1.0f + __expf(-x)); }
__device__ __forceinline__ float gelu_exact(float x) {
  return 0.5f * x * (1.0f + erff(x * 0.70710678118654752f));
}
__device__ __forceinline__ unsigned short bf16_bits(float x) {
  __bf16 h = (__bf16)x;
  return *(const unsigned short*)&h;
}

static constexpr int D    = 512;
static constexpr int Sseq = 1024;
static constexpr int Bb   = 4;

// =====================================================================
// 1) Embedding + sinusoidal positional encoding
// =====================================================================
__global__ __launch_bounds__(256) void embed_pe_kernel(
    const int* __restrict__ src, const float* __restrict__ emb,
    float* __restrict__ out) {
  const int s = blockIdx.x, b = blockIdx.y, tid = threadIdx.x;
  const int tok = src[b * Sseq + s];
  const float sq = 22.62741699796952f;  // sqrt(512)
  for (int i = 0; i < 2; ++i) {
    const int d = tid + i * 256;
    const float div = __expf(-(float)(d & ~1) * (9.210340371976184f / (float)D));
    const float ang = (float)s * div;
    const float pe  = (d & 1) ? cosf(ang) : sinf(ang);
    out[((size_t)(b * Sseq + s)) * D + d] = emb[(size_t)tok * D + d] * sq + pe;
  }
}

// =====================================================================
// 2) Depthwise conv(k=3, pad=1) + BatchNorm(eval) + ReLU
// =====================================================================
__global__ __launch_bounds__(256) void conv_bn_relu_kernel(
    const float* __restrict__ x, const float* __restrict__ cw,
    const float* __restrict__ cb, const float* __restrict__ bg,
    const float* __restrict__ bb, const float* __restrict__ bm,
    const float* __restrict__ bv, float* __restrict__ y) {
  const int s = blockIdx.x, b = blockIdx.y, tid = threadIdx.x;
  for (int i = 0; i < 2; ++i) {
    const int d = tid + i * 256;
    float acc = cb[d];
    #pragma unroll
    for (int t = 0; t < 3; ++t) {
      const int ss = s + t - 1;
      if (ss >= 0 && ss < Sseq)
        acc += x[((size_t)(b * Sseq + ss)) * D + d] * cw[d * 3 + t];
    }
    acc = (acc - bm[d]) * rsqrtf(bv[d] + 1e-5f) * bg[d] + bb[d];
    y[((size_t)(b * Sseq + s)) * D + d] = fmaxf(acc, 0.0f);
  }
}

// =====================================================================
// 3) f32 -> bf16 conversion (weights pre-pass; L2-resident afterwards)
// =====================================================================
__global__ __launch_bounds__(256) void f32_to_bf16_kernel(
    const float* __restrict__ in, unsigned short* __restrict__ out, int n) {
  const int i = blockIdx.x * 256 + threadIdx.x;
  if (i < n) out[i] = bf16_bits(in[i]);
}

// =====================================================================
// 4) Generic WMMA GEMM: C[M,N] = act( A[M,K](f32) * Wb[N,K](bf16)^T + bias (+resid) )
//    block = 256 threads (8 waves), block tile 128x128, K-step 32.
//    A staged via float4 loads + packed bf16 converts (ds_store_b64);
//    B staged via GLOBAL_LOAD_ASYNC_TO_LDS_B128 (ASYNCcnt) when available.
//    Output f32 (C) or bf16 (Cb) selected by non-null pointer.
// =====================================================================
__global__ __launch_bounds__(256) void gemm_wmma_kernel(
    const float* __restrict__ A, const unsigned short* __restrict__ Wb,
    const float* __restrict__ bias, const float* __restrict__ resid,
    float* __restrict__ C, unsigned short* __restrict__ Cb,
    int M, int N, int K, int act) {
  __attribute__((aligned(16))) __shared__ __bf16 lds_a[128 * 36];  // padded stride
  __attribute__((aligned(16))) __shared__ __bf16 lds_b[128 * 32];  // dense (16B chunks)

  const int tid  = threadIdx.x;
  const int wave = tid >> 5, lane = tid & 31;
  const int half = lane >> 4, l16 = lane & 15;
  const int wm = wave >> 1, wn = wave & 1;
  const int m0 = blockIdx.y * 128, n0 = blockIdx.x * 128;

  v8f acc[2][4];
  #pragma unroll
  for (int mi = 0; mi < 2; ++mi)
    #pragma unroll
    for (int ni = 0; ni < 4; ++ni) acc[mi][ni] = (v8f){};

  for (int k0 = 0; k0 < K; k0 += 32) {
    // ---- A tile: 128x32 f32 -> bf16, 4 float4 chunks per thread ----
    #pragma unroll
    for (int i = 0; i < 4; ++i) {
      const int c = tid + i * 256;           // 1024 chunks of 4 floats
      const int r = c >> 3, kq = (c & 7) << 2;
      const float4 f = *(const float4*)&A[(size_t)(m0 + r) * K + (k0 + kq)];
      v4bf p = {(__bf16)f.x, (__bf16)f.y, (__bf16)f.z, (__bf16)f.w};
      *(v4bf*)&lds_a[r * 36 + kq] = p;
    }
    // ---- B tile: 128x32 bf16, 2 x 16B chunks per thread ----
    #pragma unroll
    for (int i = 0; i < 2; ++i) {
      const int c = tid + i * 256;           // 512 chunks of 16B
      const int r = c >> 1, kq = (c & 1) << 4;
      const unsigned short* gp = &Wb[(size_t)(n0 + r) * K + (k0 + kq)];
      __bf16* lp = &lds_b[r * 32 + kq];
#if USE_ASYNC_LDS
      __builtin_amdgcn_global_load_async_to_lds_b128(
          (__attribute__((address_space(1))) i32x4*)gp,
          (__attribute__((address_space(3))) i32x4*)lp, 0, 0);
#else
      *(uint4*)lp = *(const uint4*)gp;
#endif
    }
    if (k0 + 32 < K) {  // gfx1250 global_prefetch_b8 of next K tiles
      __builtin_prefetch(&A[(size_t)(m0 + (tid & 127)) * K + (k0 + 32)], 0, 1);
      __builtin_prefetch(&Wb[(size_t)(n0 + (tid & 127)) * K + (k0 + 32)], 0, 1);
    }
#if USE_ASYNC_LDS
    wait_async_lds();
#endif
    __syncthreads();

    v16bf afrag[2], bfrag[4];
    #pragma unroll
    for (int mi = 0; mi < 2; ++mi) {
      const int m = wm * 32 + mi * 16 + l16;
      #pragma unroll
      for (int e = 0; e < 16; ++e) afrag[mi][e] = lds_a[m * 36 + a_kidx(e, half)];
    }
    #pragma unroll
    for (int ni = 0; ni < 4; ++ni) {
      const int n = wn * 64 + ni * 16 + l16;
      #pragma unroll
      for (int e = 0; e < 16; ++e) bfrag[ni][e] = lds_b[n * 32 + b_kidx(e, half)];
    }
    #pragma unroll
    for (int mi = 0; mi < 2; ++mi)
      #pragma unroll
      for (int ni = 0; ni < 4; ++ni)
        acc[mi][ni] = wmma_bf16(afrag[mi], bfrag[ni], acc[mi][ni]);
    __syncthreads();
  }

  // epilogue: C/D layout f32 16x16 -> row m = half*8+v, col n = l16
  #pragma unroll
  for (int mi = 0; mi < 2; ++mi) {
    #pragma unroll
    for (int ni = 0; ni < 4; ++ni) {
      #pragma unroll
      for (int v = 0; v < 8; ++v) {
        const int m = m0 + wm * 32 + mi * 16 + half * 8 + v;
        const int n = n0 + wn * 64 + ni * 16 + l16;
        float x = acc[mi][ni][v] + bias[n];
        if (resid) x += resid[(size_t)m * N + n];
        if (act == 1) x = gelu_exact(x);
        if (Cb) Cb[(size_t)m * N + n] = bf16_bits(x);
        else    C[(size_t)m * N + n] = x;
      }
    }
  }
}

// =====================================================================
// 5) LSTM recurrent kernel (single workgroup, persistent).
//    Per step: gbuf = h @ W_hh^T via WMMA (M padded 4->16), then gate math.
// =====================================================================
__global__ __launch_bounds__(256) void lstm_wmma_kernel(
    const float* __restrict__ gatesx, const unsigned short* __restrict__ whh_bits,
    const float* __restrict__ bhh, const float* __restrict__ xin,
    float* __restrict__ xout) {
  const __bf16* whh = (const __bf16*)whh_bits;  // [2048, 512]
  __shared__ float hbuf[Bb * D];
  __shared__ float cbuf[Bb * D];
  __shared__ float gbuf[Bb * 2048];

  const int tid  = threadIdx.x;
  const int wave = tid >> 5, lane = tid & 31;
  const int half = lane >> 4, l16 = lane & 15;

  for (int i = tid; i < Bb * D; i += 256) { hbuf[i] = 0.0f; cbuf[i] = 0.0f; }
  __syncthreads();

  for (int s = 0; s < Sseq; ++s) {
    #pragma unroll
    for (int grp = 0; grp < 4; ++grp) {
      v8f acc[4];
      #pragma unroll
      for (int j = 0; j < 4; ++j) acc[j] = (v8f){};
      for (int ks = 0; ks < 16; ++ks) {
        v16bf a;
        const int m = l16;
        #pragma unroll
        for (int e = 0; e < 16; ++e) {
          const int kk = ks * 32 + a_kidx(e, half);
          a[e] = (m < 4) ? (__bf16)hbuf[m * D + kk] : (__bf16)0.0f;
        }
        #pragma unroll
        for (int j = 0; j < 4; ++j) {
          const int n = (wave * 16 + grp * 4 + j) * 16 + l16;
          v16bf b;
          #pragma unroll
          for (int e = 0; e < 16; ++e)
            b[e] = whh[(size_t)n * D + (ks * 32 + b_kidx(e, half))];
          acc[j] = wmma_bf16(a, b, acc[j]);
        }
      }
      if (half == 0) {  // rows 0..3 valid (batch)
        #pragma unroll
        for (int j = 0; j < 4; ++j) {
          const int n = (wave * 16 + grp * 4 + j) * 16 + l16;
          #pragma unroll
          for (int v = 0; v < 4; ++v) gbuf[v * 2048 + n] = acc[j][v];
        }
      }
    }
    __syncthreads();

    #pragma unroll
    for (int i = 0; i < 8; ++i) {
      const int cell = tid + i * 256;
      const int b = cell >> 9, j = cell & 511;
      const size_t gb = ((size_t)(b * Sseq + s)) * 2048;
      const float gi = gatesx[gb + j]        + bhh[j]        + gbuf[b * 2048 + j];
      const float gf = gatesx[gb + 512 + j]  + bhh[512 + j]  + gbuf[b * 2048 + 512 + j];
      const float gg = gatesx[gb + 1024 + j] + bhh[1024 + j] + gbuf[b * 2048 + 1024 + j];
      const float go = gatesx[gb + 1536 + j] + bhh[1536 + j] + gbuf[b * 2048 + 1536 + j];
      const float c = sigmoidf_(gf) * cbuf[b * D + j] + sigmoidf_(gi) * tanhf(gg);
      const float h = sigmoidf_(go) * tanhf(c);
      cbuf[b * D + j] = c;
      hbuf[b * D + j] = h;
      const size_t xi = ((size_t)(b * Sseq + s)) * D + j;
      xout[xi] = xin[xi] + h;
    }
    __syncthreads();
  }
}

// =====================================================================
// 6) Attention over bf16 qkv: one block = (b, 16-query tile), 128 thr.
// =====================================================================
__global__ __launch_bounds__(128) void attn_kernel(
    const unsigned short* __restrict__ qkv_bits, float* __restrict__ ctx,
    float* __restrict__ attn) {
  const __bf16* qv = (const __bf16*)qkv_bits;  // [4096, 1536] bf16
  __shared__ __bf16 sc[16 * 1024];  // 32 KB scores
  __shared__ float red[16 * 8];

  const int tid  = threadIdx.x;
  const int wave = tid >> 5, lane = tid & 31;
  const int half = lane >> 4, l16 = lane & 15;
  const int q0 = blockIdx.x * 16, b = blockIdx.y;
  const float scale = 0.125f;  // 1/sqrt(64)

  for (int h = 0; h < 8; ++h) {
    // ---- scores = (q*scale) @ k^T ----
    for (int j = 0; j < 16; ++j) {
      const int nt = wave + 4 * j;
      v8f acc = (v8f){};
      #pragma unroll
      for (int ks = 0; ks < 2; ++ks) {
        v16bf a, bf;
        const int m = l16;
        #pragma unroll
        for (int e = 0; e < 16; ++e) {
          const int kk = ks * 32 + a_kidx(e, half);
          a[e] = (__bf16)((float)qv[((size_t)(b * Sseq + q0 + m)) * 1536 + h * 64 + kk] * scale);
        }
        const int key = nt * 16 + l16;
        #pragma unroll
        for (int e = 0; e < 16; ++e) {
          const int kk = ks * 32 + b_kidx(e, half);
          bf[e] = qv[((size_t)(b * Sseq + key)) * 1536 + 512 + h * 64 + kk];
        }
        acc = wmma_bf16(a, bf, acc);
      }
      #pragma unroll
      for (int v = 0; v < 8; ++v)
        sc[(half * 8 + v) * 1024 + nt * 16 + l16] = (__bf16)acc[v];
    }
    __syncthreads();

    // ---- softmax over 1024 keys; 8 threads per query row ----
    const int r = tid >> 3, g = tid & 7;
    float mx = -1e30f;
    for (int j = 0; j < 128; ++j)
      mx = fmaxf(mx, (float)sc[r * 1024 + g * 128 + j]);
    red[r * 8 + g] = mx;
    __syncthreads();
    float rowmax = red[r * 8];
    #pragma unroll
    for (int i = 1; i < 8; ++i) rowmax = fmaxf(rowmax, red[r * 8 + i]);
    __syncthreads();
    float sum = 0.0f;
    for (int j = 0; j < 128; ++j) {
      const int c = r * 1024 + g * 128 + j;
      const float ev = __expf((float)sc[c] - rowmax);
      sc[c] = (__bf16)ev;
      sum += ev;
    }
    red[r * 8 + g] = sum;
    __syncthreads();
    float rowsum = 0.0f;
    #pragma unroll
    for (int i = 0; i < 8; ++i) rowsum += red[r * 8 + i];
    const float inv = 1.0f / rowsum;
    for (int j = 0; j < 128; ++j) {
      const int col = g * 128 + j;
      const int c = r * 1024 + col;
      const float av = (float)sc[c] * inv;
      sc[c] = (__bf16)av;
      float* ap = attn + ((size_t)(b * Sseq + q0 + r)) * 1024 + col;
      if (h == 0) *ap = av * 0.125f;   // mean over 8 heads
      else        *ap += av * 0.125f;
    }
    __syncthreads();

    // ---- ctx = A[16,1024] @ v[1024,64] ----
    {
      v8f acc = (v8f){};
      for (int ks = 0; ks < 32; ++ks) {
        v16bf a, bf;
        #pragma unroll
        for (int e = 0; e < 16; ++e)
          a[e] = sc[l16 * 1024 + ks * 32 + a_kidx(e, half)];
        const int dim = wave * 16 + l16;
        #pragma unroll
        for (int e = 0; e < 16; ++e) {
          const int key = ks * 32 + b_kidx(e, half);
          bf[e] = qv[((size_t)(b * Sseq + key)) * 1536 + 1024 + h * 64 + dim];
        }
        acc = wmma_bf16(a, bf, acc);
      }
      #pragma unroll
      for (int v = 0; v < 8; ++v)
        ctx[((size_t)(b * Sseq + q0 + half * 8 + v)) * D + h * 64 + wave * 16 + l16] =
            acc[v];
    }
    __syncthreads();
  }
}

// =====================================================================
// 7) LayerNorm over D=512 (one block per row)
// =====================================================================
__global__ __launch_bounds__(256) void ln_kernel(
    const float* __restrict__ src, const float* __restrict__ g,
    const float* __restrict__ bt, float* __restrict__ dst) {
  __shared__ float red[256];
  const int row = blockIdx.x, tid = threadIdx.x;
  const float x0 = src[(size_t)row * D + tid];
  const float x1 = src[(size_t)row * D + 256 + tid];
  red[tid] = x0 + x1;
  __syncthreads();
  for (int s = 128; s > 0; s >>= 1) {
    if (tid < s) red[tid] += red[tid + s];
    __syncthreads();
  }
  const float mean = red[0] * (1.0f / 512.0f);
  __syncthreads();
  const float d0 = x0 - mean, d1 = x1 - mean;
  red[tid] = d0 * d0 + d1 * d1;
  __syncthreads();
  for (int s = 128; s > 0; s >>= 1) {
    if (tid < s) red[tid] += red[tid + s];
    __syncthreads();
  }
  const float inv = rsqrtf(red[0] * (1.0f / 512.0f) + 1e-5f);
  dst[(size_t)row * D + tid]       = d0 * inv * g[tid] + bt[tid];
  dst[(size_t)row * D + 256 + tid] = d1 * inv * g[256 + tid] + bt[256 + tid];
}

// =====================================================================
// Host launcher
// =====================================================================
extern "C" void kernel_launch(void* const* d_in, const int* in_sizes, int n_in,
                              void* d_out, int out_size, void* d_ws, size_t ws_size,
                              hipStream_t stream) {
  const int*   src    = (const int*)  d_in[0];
  const float* emb    = (const float*)d_in[1];
  const float* conv_w = (const float*)d_in[2];
  const float* conv_b = (const float*)d_in[3];
  const float* bn_g   = (const float*)d_in[4];
  const float* bn_b   = (const float*)d_in[5];
  const float* bn_m   = (const float*)d_in[6];
  const float* bn_v   = (const float*)d_in[7];
  const float* w_ih   = (const float*)d_in[8];
  const float* w_hh   = (const float*)d_in[9];
  const float* b_ih   = (const float*)d_in[10];
  const float* b_hh   = (const float*)d_in[11];
  const float* in_w   = (const float*)d_in[12];
  const float* in_b   = (const float*)d_in[13];
  const float* out_w  = (const float*)d_in[14];
  const float* out_b  = (const float*)d_in[15];
  const float* ln1_g  = (const float*)d_in[16];
  const float* ln1_b  = (const float*)d_in[17];
  const float* w1     = (const float*)d_in[18];
  const float* b1     = (const float*)d_in[19];
  const float* w2     = (const float*)d_in[20];
  const float* b2     = (const float*)d_in[21];
  const float* ln2_g  = (const float*)d_in[22];
  const float* ln2_b  = (const float*)d_in[23];
  const float* fin_g  = (const float*)d_in[24];
  const float* fin_b  = (const float*)d_in[25];

  // ---- workspace carve-up ----
  float* buf0 = (float*)d_ws;                  // [4096,512]  layer input x
  float* buf1 = buf0 + (size_t)4096 * 512;     // [4096,512]  conv output x2
  float* big  = buf1 + (size_t)4096 * 512;     // [4096,2048] gates_x / ff hidden
  float* ctxb = big  + (size_t)4096 * 2048;    // [4096,512]
  float* ybuf = ctxb + (size_t)4096 * 512;     // [4096,512]
  unsigned short* qkvbf = (unsigned short*)(ybuf + (size_t)4096 * 512);  // [4096,1536] bf16
  unsigned short* wb_ih = qkvbf + (size_t)4096 * 1536;   // 2048*512
  unsigned short* wb_hh = wb_ih + (size_t)2048 * 512;    // 2048*512
  unsigned short* wb_in = wb_hh + (size_t)2048 * 512;    // 4*1536*512
  unsigned short* wb_out= wb_in + (size_t)4 * 1536 * 512; // 4*512*512
  unsigned short* wb_1  = wb_out+ (size_t)4 * 512 * 512;  // 4*2048*512
  unsigned short* wb_2  = wb_1  + (size_t)4 * 2048 * 512; // 4*512*2048

  float* out_final = (float*)d_out;                      // [4,1024,512]
  float* attn_out  = out_final + (size_t)Bb * Sseq * D;  // [4,4,1024,1024]

  auto cvt = [&](const float* s, unsigned short* dptr, size_t n) {
    f32_to_bf16_kernel<<<(int)((n + 255) / 256), 256, 0, stream>>>(s, dptr, (int)n);
  };
  // weight pre-conversion to bf16 (once per launch; deterministic)
  cvt(w_ih, wb_ih, (size_t)2048 * 512);
  cvt(w_hh, wb_hh, (size_t)2048 * 512);
  cvt(in_w, wb_in, (size_t)4 * 1536 * 512);
  cvt(out_w, wb_out, (size_t)4 * 512 * 512);
  cvt(w1, wb_1, (size_t)4 * 2048 * 512);
  cvt(w2, wb_2, (size_t)4 * 512 * 2048);

  // 1) embed + PE -> buf0
  embed_pe_kernel<<<dim3(Sseq, Bb), 256, 0, stream>>>(src, emb, buf0);
  // 2) conv + BN + ReLU -> buf1
  conv_bn_relu_kernel<<<dim3(Sseq, Bb), 256, 0, stream>>>(
      buf0, conv_w, conv_b, bn_g, bn_b, bn_m, bn_v, buf1);
  // 3) gates_x = x2 @ W_ih^T + b_ih -> big
  gemm_wmma_kernel<<<dim3(2048 / 128, 4096 / 128), 256, 0, stream>>>(
      buf1, wb_ih, b_ih, nullptr, big, nullptr, 4096, 2048, 512, 0);
  // 4) LSTM recurrence; buf0 = x2 + hs
  lstm_wmma_kernel<<<1, 256, 0, stream>>>(big, wb_hh, b_hh, buf1, buf0);

  for (int l = 0; l < 4; ++l) {
    // qkv (bf16 out) = x @ in_w[l]^T + in_b[l]
    gemm_wmma_kernel<<<dim3(1536 / 128, 4096 / 128), 256, 0, stream>>>(
        buf0, wb_in + (size_t)l * 1536 * 512, in_b + (size_t)l * 1536,
        nullptr, nullptr, qkvbf, 4096, 1536, 512, 0);
    // attention: ctx + averaged attn -> d_out slab
    attn_kernel<<<dim3(Sseq / 16, Bb), 128, 0, stream>>>(
        qkvbf, ctxb, attn_out + (size_t)l * Bb * Sseq * Sseq);
    // y = x + ctx @ out_w[l]^T + out_b[l]
    gemm_wmma_kernel<<<dim3(512 / 128, 4096 / 128), 256, 0, stream>>>(
        ctxb, wb_out + (size_t)l * 512 * 512, out_b + (size_t)l * 512,
        buf0, ybuf, nullptr, 4096, 512, 512, 0);
    // LN1 -> buf0
    ln_kernel<<<4096, 256, 0, stream>>>(ybuf, ln1_g + (size_t)l * 512,
                                        ln1_b + (size_t)l * 512, buf0);
    // ff = gelu(buf0 @ w1^T + b1) -> big
    gemm_wmma_kernel<<<dim3(2048 / 128, 4096 / 128), 256, 0, stream>>>(
        buf0, wb_1 + (size_t)l * 2048 * 512, b1 + (size_t)l * 2048,
        nullptr, big, nullptr, 4096, 2048, 512, 1);
    // y = buf0 + ff @ w2^T + b2
    gemm_wmma_kernel<<<dim3(512 / 128, 4096 / 128), 256, 0, stream>>>(
        big, wb_2 + (size_t)l * 512 * 2048, b2 + (size_t)l * 512,
        buf0, ybuf, nullptr, 4096, 512, 2048, 0);
    // LN2 -> buf0
    ln_kernel<<<4096, 256, 0, stream>>>(ybuf, ln2_g + (size_t)l * 512,
                                        ln2_b + (size_t)l * 512, buf0);
  }
  // final LN -> d_out
  ln_kernel<<<4096, 256, 0, stream>>>(buf0, fin_g, fin_b, out_final);
}
